// DAGNNencoder_41601053229539
// MI455X (gfx1250) — matrix operand order repaired and checked by reading, not compiled, not run on hardware
//
#include <hip/hip_runtime.h>
#include <hip/hip_bf16.h>

// ---------------------------------------------------------------------------
// DAG-GNN encoder for MI455X (gfx1250, wave32, WMMA).
//   B=8, N=256, DE=768, DH=192, L=2, DC=192
// f16-native data path:
//   * All GEMM operands pre-converted/pre-transposed to f16 so every WMMA
//     fragment is two contiguous b128 loads per lane (no inline cvt/gather).
//   * Serial scan: V0/V1 as interleaved f16 pairs (one dword per past
//     position), GRU/wr weights packed as multiplier-grouped f16 pairs.
//   * global_prefetch of next-step rows to hide serial-chain latency.
// ---------------------------------------------------------------------------

#define Bc   8
#define Nc   256
#define DEc  768
#define DHc  192
#define Lc   2
#define DCc  192
#define G3   (3 * DHc)            // 576
#define ROWS (Bc * Nc)            // 2048
#define GRU_SZ (DHc * G3)         // 110592 per layer
#define WR_SZ  (DHc * DHc)        // 36864 per layer

typedef __attribute__((ext_vector_type(16))) _Float16 v16h;
typedef __attribute__((ext_vector_type(8)))  _Float16 v8h;
typedef __attribute__((ext_vector_type(2)))  _Float16 h2;
typedef __attribute__((ext_vector_type(8)))  float    v8f;

// ---------------- WMMA fragment helpers (wave32, 16x16x32 f16) -------------

__device__ __forceinline__ v8f wmma_f16(v16h a, v16h b, v8f c) {
  // 8 args: (neg_a, A, neg_b, B, c_mod, C, reuse_a, reuse_b)
  return __builtin_amdgcn_wmma_f32_16x16x32_f16(false, a, false, b,
                                                (short)0, c, false, false);
}

// A fragment 16x32 (MxK) from f16 row-major [ld]; lane=(16*half+m),
// element e -> K = (e>=8?16:0) + 8*half + (e%8): two contiguous v8h loads.
__device__ __forceinline__ v16h ldA_f16(const _Float16* A, int ld,
                                        int row0, int k0) {
  int lane = threadIdx.x & 31;
  int m = lane & 15, h8 = (lane >> 4) * 8;
  const _Float16* p = A + (size_t)(row0 + m) * ld + k0 + h8;
  v8h lo = *(const v8h*)p;
  v8h hi = *(const v8h*)(p + 16);
  v16h a;
#pragma unroll
  for (int e = 0; e < 8; ++e) { a[e] = lo[e]; a[e + 8] = hi[e]; }
  return a;
}

// B fragment 32x16 (KxN) from f16 COLUMN-major BT[n*ldk + k];
// lane=(16*half+n), element e -> K = (e>=8?16:0) + 8*half + (e%8): two contiguous v8h loads.
__device__ __forceinline__ v16h ldB_f16t(const _Float16* BT, int ldk,
                                         int k0, int col0) {
  int lane = threadIdx.x & 31;
  int n = lane & 15, h16 = (lane >> 4) * 8;
  const _Float16* p = BT + (size_t)(col0 + n) * ldk + k0 + h16;
  v8h lo = *(const v8h*)p;
  v8h hi = *(const v8h*)(p + 16);
  v16h b;
#pragma unroll
  for (int e = 0; e < 8; ++e) { b[e] = lo[e]; b[e + 8] = hi[e]; }
  return b;
}

// ---------------- Prep kernels ---------------------------------------------

__global__ void cvt_f16_kernel(_Float16* __restrict__ dst,
                               const float* __restrict__ src, int n) {
  int idx = blockIdx.x * blockDim.x + threadIdx.x;
  if (idx < n) dst[idx] = (_Float16)src[idx];
}

// dst[c*R + r] = (f16) src[r*C + c]   (row-major RxC -> col-major)
__global__ void transp_f16_kernel(_Float16* __restrict__ dst,
                                  const float* __restrict__ src, int R, int C) {
  int idx = blockIdx.x * blockDim.x + threadIdx.x;
  if (idx >= R * C) return;
  int r = idx / C, c = idx % C;
  dst[c * R + r] = (_Float16)src[idx];
}

// transpose two RxC matrices into interleaved f16 pairs:
// dst[2*(c*R+r)+0]=a[r*C+c], dst[2*(c*R+r)+1]=b[r*C+c]
__global__ void transp_pack2_kernel(_Float16* __restrict__ dst,
                                    const float* __restrict__ a,
                                    const float* __restrict__ b, int R, int C) {
  int idx = blockIdx.x * blockDim.x + threadIdx.x;
  if (idx >= R * C) return;
  int r = idx / C, c = idx % C;
  dst[2 * (c * R + r) + 0] = (_Float16)a[idx];
  dst[2 * (c * R + r) + 1] = (_Float16)b[idx];
}

// ---------------- Kernel: fc1  H0 = relu(F @ W + b) ------------------------
// F16: 2048x768 f16 row-major; W1T: f16 col-major [n*768+k].
__global__ void fc1_kernel(const _Float16* __restrict__ F16,
                           const _Float16* __restrict__ W1T,
                           const float* __restrict__ bias,
                           float* __restrict__ H0,
                           _Float16* __restrict__ H0h) {
  int m0 = blockIdx.x * 16, n0 = blockIdx.y * 16;
  v8f acc = {};
  for (int k0 = 0; k0 < DEc; k0 += 32)
    acc = wmma_f16(ldA_f16(F16, DEc, m0, k0), ldB_f16t(W1T, DEc, k0, n0), acc);
  int lane = threadIdx.x & 31;
  int col = n0 + (lane & 15), h8 = (lane >> 4) * 8;
  float bv = bias[col];
#pragma unroll
  for (int r = 0; r < 8; ++r) {
    float v = fmaxf(acc[r] + bv, 0.f);
    size_t off = (size_t)(m0 + r + h8) * DHc + col;
    H0[off] = v;
    H0h[off] = (_Float16)v;
  }
}

// ---------------- Block reductions (256 threads) ---------------------------

__device__ __forceinline__ float block_sum(float v, float* red, int t) {
  red[t] = v; __syncthreads();
#pragma unroll
  for (int s = 128; s > 0; s >>= 1) {
    if (t < s) red[t] += red[t + s];
    __syncthreads();
  }
  float r = red[0]; __syncthreads();
  return r;
}

__device__ __forceinline__ float block_max(float v, float* red, int t) {
  red[t] = v; __syncthreads();
#pragma unroll
  for (int s = 128; s > 0; s >>= 1) {
    if (t < s) red[t] = fmaxf(red[t], red[t + s]);
    __syncthreads();
  }
  float r = red[0]; __syncthreads();
  return r;
}

__device__ __forceinline__ float sigm(float x) {
  return 1.f / (1.f + __expf(-x));
}

// ---------------- Kernel: persistent per-batch scan ------------------------
// grid = B workgroups of 256 threads; both layers, 256 steps each.
__global__ __launch_bounds__(256) void scan_kernel(
    const float* __restrict__ adj, const float* __restrict__ smask,
    const float* __restrict__ gat_w, const float* __restrict__ gat_b,
    const float* __restrict__ cbi2, const float* __restrict__ cbh2,
    const float* __restrict__ pbi2, const float* __restrict__ pbh2,
    const _Float16* __restrict__ gpA,   // pairs (cwih, pwhh)[k*576+j], per layer
    const _Float16* __restrict__ gpB,   // pairs (cwhh, pwih)[k*576+j], per layer
    const _Float16* __restrict__ rP,    // pairs (wr0, wr1)[k*192+d], per layer
    const float* __restrict__ H0, float* __restrict__ H1,
    float* __restrict__ H2,
    _Float16* __restrict__ H1h, _Float16* __restrict__ H2h,
    _Float16* __restrict__ VPbase,      // pairs (v0,v1)[n*192+d] per batch
    float* __restrict__ As_out) {
  const int b = blockIdx.x, t = threadIdx.x;
  h2* VP = (h2*)VPbase + (size_t)b * Nc * DHc;   // L2-resident carry state

  __shared__ float xs[DHc], Ms[DHc], Hn[DHc];
  __shared__ float Ksc[Nc], attn_s[Nc], sm_s[Nc];
  __shared__ float red[256];
  __shared__ float giC[G3], ghC[G3], giP[G3], ghP[G3];

  for (int l = 0; l < Lc; ++l) {
    const float* Hin   = (l == 0) ? H0 : H1;
    float*       Hout  = (l == 0) ? H1 : H2;
    _Float16*    Houth = (l == 0) ? H1h : H2h;
    const h2* pA = (const h2*)gpA + (size_t)l * GRU_SZ;
    const h2* pB = (const h2*)gpB + (size_t)l * GRU_SZ;
    const h2* pR = (const h2*)rP + (size_t)l * WR_SZ;
    const float* wq = gat_w + l * (2 * DHc);
    const float* wk = wq + DHc;
    const float gb  = gat_b[l];
    const float* cbi = cbi2 + l * G3;  const float* cbh = cbh2 + l * G3;
    const float* pbi = pbi2 + l * G3;  const float* pbh = pbh2 + l * G3;

    // zero carry state for this layer (one f32 store zeroes a (v0,v1) pair)
    {
      float* vz = (float*)VP;
      for (int idx = t; idx < Nc * DHc; idx += 256) vz[idx] = 0.f;
    }
    Ksc[t] = 0.f;
    __syncthreads();

    for (int i = 0; i < Nc; ++i) {
      const float* adjrow = adj   + ((size_t)b * Nc + i) * Nc;
      const float* smrow  = smask + ((size_t)b * Nc + i) * Nc;
      if (i + 1 < Nc && t == 0) {   // pull next step's rows toward the WGP
        __builtin_prefetch(adj   + ((size_t)b * Nc + i + 1) * Nc, 0, 1);
        __builtin_prefetch(smask + ((size_t)b * Nc + i + 1) * Nc, 0, 1);
        __builtin_prefetch(Hin   + ((size_t)b * Nc + i + 1) * DHc, 0, 1);
      }
      if (t < DHc) xs[t] = Hin[((size_t)b * Nc + i) * DHc + t];
      sm_s[t] = smrow[t];
      __syncthreads();

      // s_q = x_i . wq (scalar broadcast)
      float sq = block_sum(t < DHc ? xs[t] * wq[t] : 0.f, red, t);

      // masked softmax over past positions (thread t == column n)
      float sc    = sq + Ksc[t] + gb;
      bool  msk   = (t < i) && (adjrow[t] > 0.5f);
      float alpha = msk ? sc : sc - 1e30f;
      float amax  = block_max(alpha, red, t);
      float e     = __expf(alpha - amax);     // unmasked -> exactly 0
      float asum  = block_sum(e, red, t);
      float at    = e / asum;
      attn_s[t]   = at;
      As_out[(((size_t)b * Lc + l) * Nc + i) * Nc + t] = (i > 0) ? at : 0.f;
      __syncthreads();

      // M[d] = sum_{n<i} attn[n] * (v0*sm + v1*(1-sm)); one dword per n
      if (t < DHc) {
        float acc = 0.f;
        for (int n2 = 0; n2 < i; ++n2) {
          h2 v = VP[n2 * DHc + t];
          float a2 = attn_s[n2], s2 = sm_s[n2];
          acc += a2 * ((float)v[0] * s2 + (float)v[1] * (1.f - s2));
        }
        Ms[t] = acc;                          // i==0 -> 0 (matches reference)
      }
      __syncthreads();

      // GRU gate matvecs: pair loads grouped by multiplier (x vs M)
      for (int j = t; j < G3; j += 256) {
        float a0 = cbi[j], a1 = cbh[j], a2 = pbi[j], a3 = pbh[j];
        for (int k = 0; k < DHc; ++k) {
          float xk = xs[k], mk = Ms[k];
          h2 wa = pA[k * G3 + j];             // (cwih[j][k], pwhh[j][k])
          h2 wb = pB[k * G3 + j];             // (cwhh[j][k], pwih[j][k])
          a0 += xk * (float)wa[0];
          a3 += xk * (float)wa[1];
          a1 += mk * (float)wb[0];
          a2 += mk * (float)wb[1];
        }
        giC[j] = a0; ghC[j] = a1; giP[j] = a2; ghP[j] = a3;
      }
      __syncthreads();

      // gate nonlinearity, Hnew = C + P
      if (t < DHc) {
        float r  = sigm(giC[t] + ghC[t]);
        float z  = sigm(giC[t + DHc] + ghC[t + DHc]);
        float nn = tanhf(giC[t + 2 * DHc] + r * ghC[t + 2 * DHc]);
        float C  = (1.f - z) * nn + z * Ms[t];
        float rp = sigm(giP[t] + ghP[t]);
        float zp = sigm(giP[t + DHc] + ghP[t + DHc]);
        float np = tanhf(giP[t + 2 * DHc] + rp * ghP[t + 2 * DHc]);
        float P  = (1.f - zp) * np + zp * xs[t];
        float hv = C + P;
        Hn[t] = hv;
        size_t off = ((size_t)b * Nc + i) * DHc + t;
        Hout[off]  = hv;
        Houth[off] = (_Float16)hv;
      }
      __syncthreads();

      // carry updates: (V0,V1)[i] = H @ (wr0,wr1)^T, Ksc[i] = H . wk
      if (t < DHc) {
        float v0 = 0.f, v1 = 0.f;
        for (int k = 0; k < DHc; ++k) {
          h2 w = pR[k * DHc + t];
          float hk = Hn[k];
          v0 += hk * (float)w[0];
          v1 += hk * (float)w[1];
        }
        h2 out; out[0] = (_Float16)v0; out[1] = (_Float16)v1;
        VP[i * DHc + t] = out;
      }
      float kres = block_sum(t < DHc ? Hn[t] * wk[t] : 0.f, red, t);
      if (t == 0) Ksc[i] = kres;
      __syncthreads();
    }
    __syncthreads();
  }
}

// ---------------- Kernel: MLP head (3 chained WMMA GEMMs) ------------------
// 4 waves per block share one staged 16-row tile; wave w does tiles w,w+4,w+8.
__global__ __launch_bounds__(128) void mlp_kernel(
    const _Float16* __restrict__ H0h, const _Float16* __restrict__ H1h,
    const _Float16* __restrict__ H2h,
    const _Float16* __restrict__ w0T, const float* __restrict__ b0,
    const _Float16* __restrict__ w1T, const float* __restrict__ b1,
    const _Float16* __restrict__ w2T, const float* __restrict__ b2,
    float* __restrict__ logits) {
  __shared__ alignas(16) _Float16 hA[16 * G3];   // concat(H0,H1,H2) tile
  __shared__ alignas(16) _Float16 hT[16 * DHc];
  __shared__ alignas(16) _Float16 hU[16 * DHc];
  const int m0 = blockIdx.x * 16;
  const int t = threadIdx.x, wave = t >> 5, lane = t & 31;
  const int n = lane & 15, h8 = (lane >> 4) * 8;

  // stage concat tile: pure v8h copies (24 chunks of 8 halves per segment)
  for (int ch = t; ch < 16 * G3 / 8; ch += 128) {
    int rr = ch / (G3 / 8);
    int cc = ch % (G3 / 8);
    int seg = cc / (DHc / 8);
    int off = (cc % (DHc / 8)) * 8;
    const _Float16* src = (seg == 0 ? H0h : (seg == 1 ? H1h : H2h)) +
                          (size_t)(m0 + rr) * DHc + off;
    *(v8h*)&hA[rr * G3 + cc * 8] = *(const v8h*)src;
  }
  __syncthreads();

  // layer 0: 16x576 @ 576x192, relu
  for (int nt = wave; nt < DHc / 16; nt += 4) {
    v8f acc = {};
    for (int k0 = 0; k0 < G3; k0 += 32)
      acc = wmma_f16(ldA_f16(hA, G3, 0, k0), ldB_f16t(w0T, G3, k0, nt * 16), acc);
    int col = nt * 16 + n; float bv = b0[col];
#pragma unroll
    for (int r = 0; r < 8; ++r)
      hT[(r + h8) * DHc + col] = (_Float16)fmaxf(acc[r] + bv, 0.f);
  }
  __syncthreads();

  // layer 1: 16x192 @ 192x192, relu
  for (int nt = wave; nt < DHc / 16; nt += 4) {
    v8f acc = {};
    for (int k0 = 0; k0 < DHc; k0 += 32)
      acc = wmma_f16(ldA_f16(hT, DHc, 0, k0), ldB_f16t(w1T, DHc, k0, nt * 16), acc);
    int col = nt * 16 + n; float bv = b1[col];
#pragma unroll
    for (int r = 0; r < 8; ++r)
      hU[(r + h8) * DHc + col] = (_Float16)fmaxf(acc[r] + bv, 0.f);
  }
  __syncthreads();

  // layer 2: 16x192 @ 192x192 -> logits
  for (int nt = wave; nt < DCc / 16; nt += 4) {
    v8f acc = {};
    for (int k0 = 0; k0 < DHc; k0 += 32)
      acc = wmma_f16(ldA_f16(hU, DHc, 0, k0), ldB_f16t(w2T, DHc, k0, nt * 16), acc);
    int col = nt * 16 + n; float bv = b2[col];
#pragma unroll
    for (int r = 0; r < 8; ++r)
      logits[(size_t)(m0 + r + h8) * DCc + col] = acc[r] + bv;
  }
}

// ---------------- Host launcher --------------------------------------------

extern "C" void kernel_launch(void* const* d_in, const int* in_sizes, int n_in,
                              void* d_out, int out_size, void* d_ws,
                              size_t ws_size, hipStream_t stream) {
  (void)in_sizes; (void)n_in; (void)out_size; (void)ws_size;

  const float* features = (const float*)d_in[0];
  const float* adj      = (const float*)d_in[1];
  const float* smask    = (const float*)d_in[2];
  // d_in[3] s_mask_onehot, d_in[4] lengths: unused by the reference math
  const float* fc1_w    = (const float*)d_in[5];
  const float* fc1_b    = (const float*)d_in[6];
  const float* gat_w    = (const float*)d_in[7];
  const float* gat_b    = (const float*)d_in[8];
  const float* wr0      = (const float*)d_in[9];
  const float* wr1      = (const float*)d_in[10];
  const float* gruc_wih = (const float*)d_in[11];
  const float* gruc_whh = (const float*)d_in[12];
  const float* gruc_bih = (const float*)d_in[13];
  const float* gruc_bhh = (const float*)d_in[14];
  const float* grup_wih = (const float*)d_in[15];
  const float* grup_whh = (const float*)d_in[16];
  const float* grup_bih = (const float*)d_in[17];
  const float* grup_bhh = (const float*)d_in[18];
  const float* mlp_w0   = (const float*)d_in[19];
  const float* mlp_b0   = (const float*)d_in[20];
  const float* mlp_w1   = (const float*)d_in[21];
  const float* mlp_b1   = (const float*)d_in[22];
  const float* mlp_w2   = (const float*)d_in[23];
  const float* mlp_b2   = (const float*)d_in[24];

  float* logits = (float*)d_out;                         // (8,256,192)
  float* As     = (float*)d_out + (size_t)Bc * Nc * DCc; // (8,2,256,256)

  // ---- workspace layout (~14.5 MB) ----
  float* H0 = (float*)d_ws;                      // f32 H buffers (scan input)
  float* H1 = H0 + (size_t)ROWS * DHc;
  float* H2 = H1 + (size_t)ROWS * DHc;
  _Float16* f16b = (_Float16*)(H2 + (size_t)ROWS * DHc);
  _Float16* F16 = f16b;                    f16b += (size_t)ROWS * DEc;
  _Float16* W1T = f16b;                    f16b += (size_t)DEc * DHc;
  _Float16* w0T = f16b;                    f16b += (size_t)G3 * DHc;
  _Float16* w1T = f16b;                    f16b += (size_t)DHc * DHc;
  _Float16* w2T = f16b;                    f16b += (size_t)DHc * DHc;
  _Float16* H0h = f16b;                    f16b += (size_t)ROWS * DHc;
  _Float16* H1h = f16b;                    f16b += (size_t)ROWS * DHc;
  _Float16* H2h = f16b;                    f16b += (size_t)ROWS * DHc;
  _Float16* gpA = f16b;                    f16b += (size_t)Lc * 2 * GRU_SZ;
  _Float16* gpB = f16b;                    f16b += (size_t)Lc * 2 * GRU_SZ;
  _Float16* rP  = f16b;                    f16b += (size_t)Lc * 2 * WR_SZ;
  _Float16* VP  = f16b;                    f16b += (size_t)Bc * Nc * DHc * 2;

  // ---- prep: convert + transpose/pack everything to f16 once ----
  {
    int n = ROWS * DEc;
    cvt_f16_kernel<<<(n + 255) / 256, 256, 0, stream>>>(F16, features, n);
  }
  auto T = [&](_Float16* dst, const float* src, int R, int C) {
    int n = R * C;
    transp_f16_kernel<<<(n + 255) / 256, 256, 0, stream>>>(dst, src, R, C);
  };
  auto P2 = [&](_Float16* dst, const float* a, const float* b, int R, int C) {
    int n = R * C;
    transp_pack2_kernel<<<(n + 255) / 256, 256, 0, stream>>>(dst, a, b, R, C);
  };
  T(W1T, fc1_w, DEc, DHc);
  T(w0T, mlp_w0, G3, DHc);
  T(w1T, mlp_w1, DHc, DHc);
  T(w2T, mlp_w2, DHc, DHc);
  for (int l = 0; l < Lc; ++l) {
    P2(gpA + (size_t)l * 2 * GRU_SZ, gruc_wih + (size_t)l * GRU_SZ,
       grup_whh + (size_t)l * GRU_SZ, G3, DHc);
    P2(gpB + (size_t)l * 2 * GRU_SZ, gruc_whh + (size_t)l * GRU_SZ,
       grup_wih + (size_t)l * GRU_SZ, G3, DHc);
    P2(rP + (size_t)l * 2 * WR_SZ, wr0 + (size_t)l * WR_SZ,
       wr1 + (size_t)l * WR_SZ, DHc, DHc);
  }

  // fc1: WMMA GEMM 2048x768x192 (b128 fragment loads)
  fc1_kernel<<<dim3(ROWS / 16, DHc / 16), 32, 0, stream>>>(F16, W1T, fc1_b,
                                                           H0, H0h);

  // serial scan: one persistent workgroup per batch element
  scan_kernel<<<Bc, 256, 0, stream>>>(adj, smask, gat_w, gat_b,
                                      gruc_bih, gruc_bhh, grup_bih, grup_bhh,
                                      gpA, gpB, rP, H0, H1, H2, H1h, H2h,
                                      VP, As);

  // MLP head: WMMA GEMM chain 2048 x (576->192->192->192)
  mlp_kernel<<<ROWS / 16, 128, 0, stream>>>(H0h, H1h, H2h, w0T, mlp_b0,
                                            w1T, mlp_b1, w2T, mlp_b2, logits);
}
